// MixtralSparseMoeBlockOurs_4861902979225
// MI455X (gfx1250) — matrix operand order repaired and checked by reading, not compiled, not run on hardware
//
#include <hip/hip_runtime.h>
#include <hip/hip_bf16.h>

// ---------------------------------------------------------------------------
// Mixtral sparse MoE block for gfx1250 (MI455X).
//   * bf16 WMMA (v_wmma_f32_16x16x32_bf16), f32 accumulation
//   * per-expert weights pre-converted + pre-transposed to bf16 [n][k]
//   * GEMM tiles staged by the Tensor Data Mover (tensor_load_to_lds),
//     descriptor-driven DMA with built-in LDS padding, double-buffered and
//     software-pipelined against the WMMAs (TENSORcnt / s_wait_tensorcnt).
// ---------------------------------------------------------------------------

typedef __attribute__((ext_vector_type(16))) __bf16 v16bf;
typedef __attribute__((ext_vector_type(8)))  float  v8f;
typedef __attribute__((ext_vector_type(4)))  unsigned int u32x4;
typedef __attribute__((ext_vector_type(8)))  unsigned int u32x8;

static constexpr int T = 4096;
static constexpr int H = 1024;
static constexpr int F = 3584;
static constexpr int E = 8;

static constexpr int BM  = 128;   // block tile M
static constexpr int BN  = 128;   // block tile N
static constexpr int BK  = 32;    // K per WMMA step
static constexpr int LDA = 40;    // LDS row stride (bf16): 64B data + 16B pad

__device__ __forceinline__ unsigned short f32_bf16(float f) {
  union { float f; unsigned u; } v; v.f = f;
  unsigned r = v.u + 0x7FFFu + ((v.u >> 16) & 1u);   // round-to-nearest-even
  return (unsigned short)(r >> 16);
}

union Frag { v16bf v; unsigned u[8]; };
union Acc  { v8f   v; float    f[8]; };
union G0   { unsigned w[4]; u32x4 v; };
union G1   { unsigned w[8]; u32x8 v; };

// ---- Tensor Data Mover: D# group0 (flags | lds_addr | global_addr | type) --
__device__ __forceinline__ u32x4 tdm_g0(unsigned lds_off, unsigned long long gaddr) {
  G0 g;
  g.w[0] = 1u;                                   // count=1, no gather
  g.w[1] = lds_off;                              // lds_addr
  g.w[2] = (unsigned)gaddr;                      // global_addr[31:0]
  g.w[3] = (unsigned)(gaddr >> 32) | (2u << 30); // global_addr[56:32] | type=2
  return g.v;
}
// ---- D# group1: 2D tile 32 x 128 elems (2B), pad 16 DWORDs -> +4 DWORDs ----
__device__ __forceinline__ u32x8 tdm_g1(unsigned dim0, unsigned dim1, unsigned stride0) {
  G1 g;
  g.w[0] = (1u << 16)            // data_size = 1 (2 bytes)
         | (1u << 20)            // pad_enable
         | (3u << 22)            // pad_interval: code 3 = 16 DWORDs (64B)
         | (3u << 25);           // pad_amount:   code 3 = 4 DWORDs (16B)
  g.w[1] = (dim0 & 0xFFFFu) << 16;                       // tensor_dim0[15:0]
  g.w[2] = (dim0 >> 16) | ((dim1 & 0xFFFFu) << 16);      // dim0[31:16] | dim1[15:0]
  g.w[3] = (dim1 >> 16) | (32u << 16);                   // dim1[31:16] | tile_dim0=32
  g.w[4] = 128u;                                         // tile_dim1=128, tile_dim2=0
  g.w[5] = stride0;                                      // tensor_dim0_stride[31:0]
  g.w[6] = 0u;
  g.w[7] = 0u;
  return g.v;
}
__device__ __forceinline__ void tdm_load(u32x4 g0, u32x8 g1) {
  asm volatile("tensor_load_to_lds %0, %1" :: "s"(g0), "s"(g1) : "memory");
}
__device__ __forceinline__ void wait_tensor0() {
  __builtin_amdgcn_s_wait_tensorcnt(0);
}
__device__ __forceinline__ unsigned lds_off(const void* p) {
  return (unsigned)(unsigned long long)p;   // low 32b of generic ptr = LDS offset
}

// ---------------------------------------------------------------------------
// out zero-init (we accumulate across experts)
// ---------------------------------------------------------------------------
__global__ __launch_bounds__(256) void zero_out_kernel(float* __restrict__ out) {
  int i = (blockIdx.x * 256 + threadIdx.x) * 4;
  *(float4*)(out + i) = float4{0.f, 0.f, 0.f, 0.f};
}

// ---------------------------------------------------------------------------
// f32 -> bf16 bulk convert (hidden states)
// ---------------------------------------------------------------------------
__global__ __launch_bounds__(256) void cvt_bf16_kernel(const float* __restrict__ src,
                                                       unsigned short* __restrict__ dst) {
  int i = (blockIdx.x * 256 + threadIdx.x) * 8;
  float4 a = *(const float4*)(src + i);
  float4 b = *(const float4*)(src + i + 4);
  uint4 o;
  o.x = (unsigned)f32_bf16(a.x) | ((unsigned)f32_bf16(a.y) << 16);
  o.y = (unsigned)f32_bf16(a.z) | ((unsigned)f32_bf16(a.w) << 16);
  o.z = (unsigned)f32_bf16(b.x) | ((unsigned)f32_bf16(b.y) << 16);
  o.w = (unsigned)f32_bf16(b.z) | ((unsigned)f32_bf16(b.w) << 16);
  *(uint4*)(dst + i) = o;
}

// ---------------------------------------------------------------------------
// Transpose + convert: dst[c][r] = bf16(src[r][c]).  src: R x C f32, dst: C x R bf16.
// ---------------------------------------------------------------------------
__global__ __launch_bounds__(256) void transpose_cvt_kernel(const float* __restrict__ src,
                                                            unsigned short* __restrict__ dst,
                                                            int R, int C) {
  __shared__ alignas(16) unsigned short tile[64][80];
  const int r0 = blockIdx.y * 64;
  const int c0 = blockIdx.x * 64;
  const int tid = threadIdx.x;
#pragma unroll
  for (int i = 0; i < 4; ++i) {
    int slot = tid + i * 256;
    int r = slot >> 4;
    int c = (slot & 15) * 4;
    float4 v = *(const float4*)(src + (size_t)(r0 + r) * C + c0 + c);
    tile[c + 0][r] = f32_bf16(v.x);
    tile[c + 1][r] = f32_bf16(v.y);
    tile[c + 2][r] = f32_bf16(v.z);
    tile[c + 3][r] = f32_bf16(v.w);
  }
  __syncthreads();
#pragma unroll
  for (int i = 0; i < 2; ++i) {
    int slot = tid + i * 256;
    int c = slot >> 3;
    int p = (slot & 7) * 8;
    uint4 d = *(uint4*)&tile[c][p];
    *(uint4*)(dst + (size_t)(c0 + c) * R + r0 + p) = d;
  }
}

// ---------------------------------------------------------------------------
// Router: logits = hs @ gate_w^T, softmax over E=8, top-2, renormalize.
// ---------------------------------------------------------------------------
__global__ __launch_bounds__(64) void router_kernel(const float* __restrict__ hs,
                                                    const float* __restrict__ gw,
                                                    float* __restrict__ topk_out,
                                                    float* __restrict__ expertW) {
  __shared__ float red[64 * E];
  const int t   = blockIdx.x;
  const int tid = threadIdx.x;

  float p[E];
#pragma unroll
  for (int e = 0; e < E; ++e) p[e] = 0.f;

  const float* x = hs + (size_t)t * H;
  const int h0 = tid * (H / 64);
#pragma unroll 4
  for (int h = h0; h < h0 + (H / 64); ++h) {
    float xv = x[h];
#pragma unroll
    for (int e = 0; e < E; ++e) p[e] += xv * gw[e * H + h];
  }
#pragma unroll
  for (int e = 0; e < E; ++e) red[tid * E + e] = p[e];
  __syncthreads();

  if (tid == 0) {
    float logits[E];
#pragma unroll
    for (int e = 0; e < E; ++e) {
      float s = 0.f;
      for (int i = 0; i < 64; ++i) s += red[i * E + e];
      logits[e] = s;
    }
    float mx = logits[0];
#pragma unroll
    for (int e = 1; e < E; ++e) mx = fmaxf(mx, logits[e]);
    float pr[E], se = 0.f;
#pragma unroll
    for (int e = 0; e < E; ++e) { pr[e] = __expf(logits[e] - mx); se += pr[e]; }
    float inv = 1.f / se;
#pragma unroll
    for (int e = 0; e < E; ++e) pr[e] *= inv;

    int i0 = 0;
#pragma unroll
    for (int e = 1; e < E; ++e) if (pr[e] > pr[i0]) i0 = e;
    int i1 = (i0 == 0) ? 1 : 0;
#pragma unroll
    for (int e = 0; e < E; ++e) if (e != i0 && pr[e] > pr[i1]) i1 = e;

    float s2 = pr[i0] + pr[i1];
    float w0 = pr[i0] / s2, w1 = pr[i1] / s2;
    topk_out[t * 2 + 0] = w0;
    topk_out[t * 2 + 1] = w1;
#pragma unroll
    for (int e = 0; e < E; ++e)
      expertW[t * E + e] = (e == i0) ? w0 : ((e == i1) ? w1 : 0.f);
  }
}

// ---------------------------------------------------------------------------
// GEMM1 (fused gate/up + SiLU): act = silu(A @ Wg') * (A @ Wu')
// A: T x H bf16 [m][k].  Wb: 2F x H bf16 [n][k] (up rows at +F).
// TDM-staged, double-buffered.  8 waves: 4 along M x 2 along N.
// ---------------------------------------------------------------------------
__global__ __launch_bounds__(256) void gemm1_kernel(const unsigned short* __restrict__ A,
                                                    const unsigned short* __restrict__ Wb,
                                                    unsigned short* __restrict__ act) {
  __shared__ alignas(16) unsigned short As[2][BM * LDA];
  __shared__ alignas(16) unsigned short Bg[2][BN * LDA];
  __shared__ alignas(16) unsigned short Bu[2][BN * LDA];

  const int m0   = blockIdx.y * BM;
  const int n0   = blockIdx.x * BN;
  const int tid  = threadIdx.x;
  const int lane = tid & 31;
  const int wid  = tid >> 5;
  const int wm   = wid & 3;
  const int wn   = wid >> 2;
  const int l16  = lane & 15;
  const int lh   = lane >> 4;

  const unsigned short* Wg = Wb;
  const unsigned short* Wu = Wb + (size_t)F * H;

  const u32x8 g1 = tdm_g1((unsigned)H, (unsigned)T, (unsigned)H);

  Acc accG[2][4], accU[2][4];
#pragma unroll
  for (int mt = 0; mt < 2; ++mt)
#pragma unroll
    for (int nt = 0; nt < 4; ++nt)
#pragma unroll
      for (int i = 0; i < 8; ++i) { accG[mt][nt].f[i] = 0.f; accU[mt][nt].f[i] = 0.f; }

  auto stage = [&](int buf, int k0) {
    if (wid == 0) {
      unsigned long long ga = (unsigned long long)A  + ((size_t)m0 * H + k0) * 2;
      unsigned long long gg = (unsigned long long)Wg + ((size_t)n0 * H + k0) * 2;
      unsigned long long gu = (unsigned long long)Wu + ((size_t)n0 * H + k0) * 2;
      tdm_load(tdm_g0(lds_off(&As[buf][0]), ga), g1);
      tdm_load(tdm_g0(lds_off(&Bg[buf][0]), gg), g1);
      tdm_load(tdm_g0(lds_off(&Bu[buf][0]), gu), g1);
    }
  };

  constexpr int NK = H / BK;
  stage(0, 0);
  for (int kt = 0; kt < NK; ++kt) {
    const int buf = kt & 1;
    wait_tensor0();            // no-op for waves with TENSORcnt==0
    __syncthreads();
    if (kt + 1 < NK) stage(buf ^ 1, (kt + 1) * BK);

    Frag a[2], bg[4], bu[4];
#pragma unroll
    for (int mt = 0; mt < 2; ++mt) {
      const unsigned* Ar = (const unsigned*)&As[buf][(wm * 32 + mt * 16 + l16) * LDA];
#pragma unroll
      for (int v = 0; v < 8; ++v) {
        int k = 2 * v + lh * 16;   // A 16x32 bf16: k = lh*16 + 2v
        a[mt].u[v] = Ar[k >> 1];
      }
    }
#pragma unroll
    for (int nt = 0; nt < 4; ++nt) {
      const unsigned* Brg = (const unsigned*)&Bg[buf][(wn * 64 + nt * 16 + l16) * LDA];
      const unsigned* Bru = (const unsigned*)&Bu[buf][(wn * 64 + nt * 16 + l16) * LDA];
#pragma unroll
      for (int v = 0; v < 8; ++v) {                    // B 32x16: k = lh*16 + 2v
        bg[nt].u[v] = Brg[lh * 8 + v];
        bu[nt].u[v] = Bru[lh * 8 + v];
      }
    }
#pragma unroll
    for (int mt = 0; mt < 2; ++mt)
#pragma unroll
      for (int nt = 0; nt < 4; ++nt) {
        accG[mt][nt].v = __builtin_amdgcn_wmma_f32_16x16x32_bf16(
            false, a[mt].v, false, bg[nt].v, (short)0, accG[mt][nt].v, false, false);
        accU[mt][nt].v = __builtin_amdgcn_wmma_f32_16x16x32_bf16(
            false, a[mt].v, false, bu[nt].v, (short)0, accU[mt][nt].v, false, false);
      }
    __syncthreads();
  }

  // epilogue: act = g*sigmoid(g)*u -> bf16
#pragma unroll
  for (int mt = 0; mt < 2; ++mt)
#pragma unroll
    for (int nt = 0; nt < 4; ++nt)
#pragma unroll
      for (int i = 0; i < 8; ++i) {
        float g = accG[mt][nt].f[i];
        float u = accU[mt][nt].f[i];
        float s = g * u * (1.f / (1.f + __expf(-g)));
        int mg = m0 + wm * 32 + mt * 16 + lh * 8 + i;
        int ng = n0 + wn * 64 + nt * 16 + l16;
        act[(size_t)mg * F + ng] = f32_bf16(s);
      }
}

// ---------------------------------------------------------------------------
// GEMM2: out += expert_w[:,e] * (act @ w2).  A: T x F bf16 [m][k],
// Wb: H x F bf16 [n][k].  TDM-staged.  8 waves: 2 along M x 4 along N.
// ---------------------------------------------------------------------------
__global__ __launch_bounds__(256) void gemm2_kernel(const unsigned short* __restrict__ A,
                                                    const unsigned short* __restrict__ Wb,
                                                    const float* __restrict__ expertW,
                                                    int e,
                                                    float* __restrict__ out) {
  __shared__ alignas(16) unsigned short As[2][BM * LDA];
  __shared__ alignas(16) unsigned short Bs[2][BN * LDA];

  const int m0   = blockIdx.y * BM;
  const int n0   = blockIdx.x * BN;
  const int tid  = threadIdx.x;
  const int lane = tid & 31;
  const int wid  = tid >> 5;
  const int wm   = wid & 1;
  const int wn   = wid >> 1;
  const int l16  = lane & 15;
  const int lh   = lane >> 4;

  const u32x8 g1 = tdm_g1((unsigned)F, (unsigned)T, (unsigned)F);

  Acc acc[4][2];
#pragma unroll
  for (int mt = 0; mt < 4; ++mt)
#pragma unroll
    for (int nt = 0; nt < 2; ++nt)
#pragma unroll
      for (int i = 0; i < 8; ++i) acc[mt][nt].f[i] = 0.f;

  auto stage = [&](int buf, int k0) {
    if (wid == 0) {
      unsigned long long ga = (unsigned long long)A  + ((size_t)m0 * F + k0) * 2;
      unsigned long long gb = (unsigned long long)Wb + ((size_t)n0 * F + k0) * 2;
      tdm_load(tdm_g0(lds_off(&As[buf][0]), ga), g1);
      tdm_load(tdm_g0(lds_off(&Bs[buf][0]), gb), g1);
    }
  };

  constexpr int NK = F / BK;
  stage(0, 0);
  for (int kt = 0; kt < NK; ++kt) {
    const int buf = kt & 1;
    wait_tensor0();
    __syncthreads();
    if (kt + 1 < NK) stage(buf ^ 1, (kt + 1) * BK);

    Frag a[4], b[2];
#pragma unroll
    for (int mt = 0; mt < 4; ++mt) {
      const unsigned* Ar = (const unsigned*)&As[buf][(wm * 64 + mt * 16 + l16) * LDA];
#pragma unroll
      for (int v = 0; v < 8; ++v) {
        int k = 2 * v + lh * 16;
        a[mt].u[v] = Ar[k >> 1];
      }
    }
#pragma unroll
    for (int nt = 0; nt < 2; ++nt) {
      const unsigned* Br = (const unsigned*)&Bs[buf][(wn * 32 + nt * 16 + l16) * LDA];
#pragma unroll
      for (int v = 0; v < 8; ++v) b[nt].u[v] = Br[lh * 8 + v];
    }
#pragma unroll
    for (int mt = 0; mt < 4; ++mt)
#pragma unroll
      for (int nt = 0; nt < 2; ++nt)
        acc[mt][nt].v = __builtin_amdgcn_wmma_f32_16x16x32_bf16(
            false, a[mt].v, false, b[nt].v, (short)0, acc[mt][nt].v, false, false);
    __syncthreads();
  }

  // scaled accumulate (block owns its tile; experts are stream-ordered)
#pragma unroll
  for (int mt = 0; mt < 4; ++mt)
#pragma unroll
    for (int nt = 0; nt < 2; ++nt)
#pragma unroll
      for (int i = 0; i < 8; ++i) {
        int mg = m0 + wm * 64 + mt * 16 + lh * 8 + i;
        int ng = n0 + wn * 32 + nt * 16 + l16;
        float s = expertW[mg * E + e];
        float* dst = out + (size_t)mg * H + ng;
        *dst = *dst + s * acc[mt][nt].f[i];
      }
}

// ---------------------------------------------------------------------------
// Host launcher
// ---------------------------------------------------------------------------
extern "C" void kernel_launch(void* const* d_in, const int* in_sizes, int n_in,
                              void* d_out, int out_size, void* d_ws, size_t ws_size,
                              hipStream_t stream) {
  (void)in_sizes; (void)n_in; (void)out_size; (void)ws_size;

  const float* hs  = (const float*)d_in[0];   // T x H
  const float* gw  = (const float*)d_in[1];   // E x H
  const float* w13 = (const float*)d_in[2];   // E x H x 2F
  const float* w2  = (const float*)d_in[3];   // E x F x H

  float* out  = (float*)d_out;                // T x H
  float* topk = out + (size_t)T * H;          // T x 2

  // ws layout: expert_w | hs bf16 | act bf16 | w13b bf16 (2F x H) | w2b bf16 (H x F)
  char* p = (char*)d_ws;
  float*          expertW = (float*)p;            p += (size_t)T * E * sizeof(float);
  unsigned short* hsb     = (unsigned short*)p;   p += (size_t)T * H * 2;
  unsigned short* act     = (unsigned short*)p;   p += (size_t)T * F * 2;
  unsigned short* w13b    = (unsigned short*)p;   p += (size_t)2 * F * H * 2;
  unsigned short* w2b     = (unsigned short*)p;

  zero_out_kernel<<<(T * H) / 1024, 256, 0, stream>>>(out);
  router_kernel<<<T, 64, 0, stream>>>(hs, gw, topk, expertW);
  cvt_bf16_kernel<<<(T * H) / 2048, 256, 0, stream>>>(hs, hsb);

  for (int e = 0; e < E; ++e) {
    transpose_cvt_kernel<<<dim3((2 * F) / 64, H / 64), 256, 0, stream>>>(
        w13 + (size_t)e * H * 2 * F, w13b, H, 2 * F);
    transpose_cvt_kernel<<<dim3(H / 64, F / 64), 256, 0, stream>>>(
        w2 + (size_t)e * F * H, w2b, F, H);

    gemm1_kernel<<<dim3(F / BN, T / BM), 256, 0, stream>>>(hsb, w13b, act);
    gemm2_kernel<<<dim3(H / BN, T / BM), 256, 0, stream>>>(act, w2b, expertW, e, out);
  }
}
